// PairFeaturesLayer_63496796504636
// MI455X (gfx1250) — matrix operand order repaired and checked by reading, not compiled, not run on hardware
//
#include <hip/hip_runtime.h>
#include <hip/hip_bf16.h>

typedef __attribute__((ext_vector_type(16))) _Float16 v16h;
typedef __attribute__((ext_vector_type(8)))  _Float16 v8h;
typedef __attribute__((ext_vector_type(4)))  _Float16 v4h;
typedef __attribute__((ext_vector_type(8)))  float    v8f;
typedef __attribute__((ext_vector_type(4)))  float    v4f;

#define NODE_NF 256
#define EDGE_NF 128
#define OUT_NF  256

// ---------------------------------------------------------------------------
// Pack a weight [K, N] (f32, row-major as in reference: y = x @ W) into a
// transposed f16 buffer WT[N][K] so WMMA B-fragments are contiguous per lane.
// ---------------------------------------------------------------------------
__global__ void pack_weight_T(const float* __restrict__ W, _Float16* __restrict__ WT,
                              int K, int N) {
    int tid = blockIdx.x * blockDim.x + threadIdx.x;
    int total = K * N;
    if (tid >= total) return;
    int n = tid / K;
    int k = tid - n * K;
    WT[tid] = (_Float16)W[(size_t)k * N + n];
}

// f32 -> f16 bulk convert (4 elems/thread, vectorized)
__global__ void pack_f16(const float* __restrict__ X, _Float16* __restrict__ Y, int total4) {
    int tid = blockIdx.x * blockDim.x + threadIdx.x;
    if (tid >= total4) return;
    v4f f = *(const v4f*)(X + (size_t)tid * 4);
    v4h h;
    h[0] = (_Float16)f[0]; h[1] = (_Float16)f[1];
    h[2] = (_Float16)f[2]; h[3] = (_Float16)f[3];
    *(v4h*)(Y + (size_t)tid * 4) = h;
}

__device__ __forceinline__ v16h cat8(v8h lo, v8h hi) {
    return __builtin_shufflevector(lo, hi, 0,1,2,3,4,5,6,7,8,9,10,11,12,13,14,15);
}

// Load one 16x32 f16 A-fragment from an LDS tile with row stride `ldk` halfs.
// ISA layout: lane holds M = lane%16; halfs 0..7 -> K = klo..klo+7,
// halfs 8..15 -> K = 16+klo..16+klo+7, where klo = (lane/16)*8.
__device__ __forceinline__ v16h load_a_frag(const _Float16* sRow, int kc, int klo) {
    return cat8(*(const v8h*)(sRow + kc + klo),
                *(const v8h*)(sRow + kc + 16 + klo));
}

// Load one 32x16 f16 B-fragment from a transposed weight table WT[N][Kdim].
// ISA layout: lane holds N = lane%16; halfs 0..7 -> K = kc+klo..+7,
// halfs 8..15 -> K = kc+16+klo..+7, klo = (lane/16)*8 -> two 16B loads.
__device__ __forceinline__ v16h load_b_frag(const _Float16* WT, int col, int Kdim,
                                            int kc, int khalf) {
    const _Float16* pb = WT + (size_t)col * Kdim + kc + khalf / 2;
    return cat8(*(const v8h*)pb, *(const v8h*)(pb + 16));
}

// ---------------------------------------------------------------------------
// Kernel 3: P = nf @ W_i, Q = nf @ W_j   (M=16 nodes per block, N=256, K=256)
// 256 threads = 8 waves; wave w owns N-tiles {2w, 2w+1}.
// K-chunk loop outermost: each A fragment loaded from LDS once, reused by
// 4 accumulators (2 N-tiles x {W_i, W_j}).
// ---------------------------------------------------------------------------
__global__ __launch_bounds__(256) void node_proj_kernel(
    const _Float16* __restrict__ nf16,
    const _Float16* __restrict__ WiT,   // [256][256] f16
    const _Float16* __restrict__ WjT,   // [256][256] f16
    float* __restrict__ P, float* __restrict__ Q) {
    __shared__ __align__(16) _Float16 sA[16 * NODE_NF];   // 8 KB

    int tid   = threadIdx.x;
    int mtile = blockIdx.x;

    // Stage 16 rows x 256 halfs of nf16 into LDS (16 halfs per thread)
    {
        const _Float16* src = nf16 + (size_t)mtile * 16 * NODE_NF;
        int off = tid * 16;
        *(v8h*)(sA + off)     = *(const v8h*)(src + off);
        *(v8h*)(sA + off + 8) = *(const v8h*)(src + off + 8);
    }
    __syncthreads();

    int lane  = tid & 31;
    int wave  = tid >> 5;
    int nl    = lane & 15;          // A: M row / B,C: N col (mod 16)
    int klo   = (lane >> 4) * 8;    // A-fragment K sub-offset
    int khalf = (lane >> 4) * 16;   // B-fragment K sub-offset
    int mbase = (lane >> 4) * 8;    // C-fragment M base

    int col0 = (wave * 2 + 0) * 16 + nl;
    int col1 = (wave * 2 + 1) * 16 + nl;
    const _Float16* sRow = sA + nl * NODE_NF;

    v8f accP0 = {}, accP1 = {}, accQ0 = {}, accQ1 = {};
    #pragma unroll
    for (int c = 0; c < NODE_NF / 32; ++c) {
        int kc = c * 32;
        v16h a  = load_a_frag(sRow, kc, klo);
        v16h bi0 = load_b_frag(WiT, col0, NODE_NF, kc, khalf);
        v16h bi1 = load_b_frag(WiT, col1, NODE_NF, kc, khalf);
        v16h bj0 = load_b_frag(WjT, col0, NODE_NF, kc, khalf);
        v16h bj1 = load_b_frag(WjT, col1, NODE_NF, kc, khalf);
        accP0 = __builtin_amdgcn_wmma_f32_16x16x32_f16(false, a, false, bi0, (short)0, accP0, false, false);
        accP1 = __builtin_amdgcn_wmma_f32_16x16x32_f16(false, a, false, bi1, (short)0, accP1, false, false);
        accQ0 = __builtin_amdgcn_wmma_f32_16x16x32_f16(false, a, false, bj0, (short)0, accQ0, false, false);
        accQ1 = __builtin_amdgcn_wmma_f32_16x16x32_f16(false, a, false, bj1, (short)0, accQ1, false, false);
    }

    #pragma unroll
    for (int r = 0; r < 8; ++r) {
        size_t row = (size_t)(mtile * 16 + mbase + r) * OUT_NF;
        P[row + col0] = accP0[r];
        P[row + col1] = accP1[r];
        Q[row + col0] = accQ0[r];
        Q[row + col1] = accQ1[r];
    }
}

// ---------------------------------------------------------------------------
// Kernel 4: out[e] = relu(edge_attr[e] @ W_ij + b + P[i_e] + Q[j_e])
// M=16 edges per block, N=256, K=128. Gathered P/Q (41 MB) live in L2.
// Accumulators are initialized directly with bias + gathered projections
// (per the C-matrix VGPR layout), so the gather IS the accumulate-init.
// ---------------------------------------------------------------------------
__global__ __launch_bounds__(256) void edge_kernel(
    const float* __restrict__ edge_attr,
    const int*   __restrict__ eidx,      // [2][n_edges] int32 (harness int layout)
    const float* __restrict__ b_ij,
    const float* __restrict__ P, const float* __restrict__ Q,
    const _Float16* __restrict__ WijT,   // [256][128] f16
    float* __restrict__ out, int n_edges) {
    __shared__ __align__(16) _Float16 sA[16 * EDGE_NF];   // 4 KB
    __shared__ int sIdx[32];                              // i[0..15], j[0..15]

    int tid   = threadIdx.x;
    int mtile = blockIdx.x;
    int ebase = mtile * 16;

    if (tid < 32) {
        int m     = tid & 15;
        int which = tid >> 4;
        sIdx[tid] = eidx[(size_t)which * n_edges + ebase + m];
    }
    // Stage 16 x 128 edge_attr tile, f32 -> f16 (8 elems per thread)
    {
        int idx = tid * 8;             // 0..2040
        int row = idx >> 7;
        int col = idx & 127;
        const float* src = edge_attr + (size_t)(ebase + row) * EDGE_NF + col;
        v4f f0 = *(const v4f*)src;
        v4f f1 = *(const v4f*)(src + 4);
        v8h h;
        #pragma unroll
        for (int q = 0; q < 4; ++q) {
            h[q]     = (_Float16)f0[q];
            h[4 + q] = (_Float16)f1[q];
        }
        *(v8h*)(sA + idx) = h;
    }
    __syncthreads();

    int lane  = tid & 31;
    int wave  = tid >> 5;
    int nl    = lane & 15;
    int klo   = (lane >> 4) * 8;
    int khalf = (lane >> 4) * 16;
    int mbase = (lane >> 4) * 8;

    int col0 = (wave * 2 + 0) * 16 + nl;
    int col1 = (wave * 2 + 1) * 16 + nl;
    const _Float16* sRow = sA + nl * EDGE_NF;

    // Accumulator init = bias + gathered node projections (C layout).
    // P/Q tables (41 MB) are L2-resident; these are L2 hits.
    float bias0 = b_ij[col0];
    float bias1 = b_ij[col1];
    v8f acc0, acc1;
    #pragma unroll
    for (int r = 0; r < 8; ++r) {
        int m  = mbase + r;
        size_t ri = (size_t)sIdx[m]      * OUT_NF;
        size_t rj = (size_t)sIdx[16 + m] * OUT_NF;
        acc0[r] = bias0 + P[ri + col0] + Q[rj + col0];
        acc1[r] = bias1 + P[ri + col1] + Q[rj + col1];
    }

    // Edge GEMM: 4 K-chunks of 32; A fragment loaded once per chunk.
    #pragma unroll
    for (int c = 0; c < EDGE_NF / 32; ++c) {
        int kc = c * 32;
        v16h a  = load_a_frag(sRow, kc, klo);
        v16h b0 = load_b_frag(WijT, col0, EDGE_NF, kc, khalf);
        v16h b1 = load_b_frag(WijT, col1, EDGE_NF, kc, khalf);
        acc0 = __builtin_amdgcn_wmma_f32_16x16x32_f16(false, a, false, b0, (short)0, acc0, false, false);
        acc1 = __builtin_amdgcn_wmma_f32_16x16x32_f16(false, a, false, b1, (short)0, acc1, false, false);
    }

    // ReLU + store
    #pragma unroll
    for (int r = 0; r < 8; ++r) {
        size_t row = (size_t)(ebase + mbase + r) * OUT_NF;
        float v0 = acc0[r];
        float v1 = acc1[r];
        out[row + col0] = v0 > 0.0f ? v0 : 0.0f;
        out[row + col1] = v1 > 0.0f ? v1 : 0.0f;
    }
}

extern "C" void kernel_launch(void* const* d_in, const int* in_sizes, int n_in,
                              void* d_out, int out_size, void* d_ws, size_t ws_size,
                              hipStream_t stream) {
    const float* node_feats = (const float*)d_in[0];
    const int*   edge_index = (const int*)d_in[1];   // per harness: integer -> int*
    const float* edge_attr  = (const float*)d_in[2];
    const float* W_ij       = (const float*)d_in[3];
    const float* b_ij       = (const float*)d_in[4];
    const float* W_i        = (const float*)d_in[5];
    const float* W_j        = (const float*)d_in[6];
    float*       out        = (float*)d_out;

    const int n_nodes = in_sizes[0] / NODE_NF;   // 20000
    const int n_edges = in_sizes[1] / 2;         // 640000

    // Workspace layout (all 256B-aligned):
    //   P     [n_nodes][256] f32   20.5 MB
    //   Q     [n_nodes][256] f32   20.5 MB
    //   WijT  [256][128]     f16   64 KB
    //   WiT   [256][256]     f16   128 KB
    //   WjT   [256][256]     f16   128 KB
    //   nf16  [n_nodes][256] f16   10.2 MB
    char* ws = (char*)d_ws;
    auto align256 = [](size_t x) { return (x + 255) & ~(size_t)255; };
    float* P = (float*)ws;                 ws += align256((size_t)n_nodes * OUT_NF * 4);
    float* Q = (float*)ws;                 ws += align256((size_t)n_nodes * OUT_NF * 4);
    _Float16* WijT = (_Float16*)ws;        ws += align256((size_t)OUT_NF * EDGE_NF * 2);
    _Float16* WiT  = (_Float16*)ws;        ws += align256((size_t)OUT_NF * NODE_NF * 2);
    _Float16* WjT  = (_Float16*)ws;        ws += align256((size_t)OUT_NF * NODE_NF * 2);
    _Float16* nf16 = (_Float16*)ws;

    // 1) Pack weights (transposed f16)
    pack_weight_T<<<(EDGE_NF * OUT_NF + 255) / 256, 256, 0, stream>>>(W_ij, WijT, EDGE_NF, OUT_NF);
    pack_weight_T<<<(NODE_NF * OUT_NF + 255) / 256, 256, 0, stream>>>(W_i,  WiT,  NODE_NF, OUT_NF);
    pack_weight_T<<<(NODE_NF * OUT_NF + 255) / 256, 256, 0, stream>>>(W_j,  WjT,  NODE_NF, OUT_NF);

    // 2) node_feats -> f16
    {
        int total4 = (n_nodes * NODE_NF) / 4;
        pack_f16<<<(total4 + 255) / 256, 256, 0, stream>>>(node_feats, nf16, total4);
    }

    // 3) Node projections P, Q (one 16-row tile per block)
    node_proj_kernel<<<n_nodes / 16, 256, 0, stream>>>(nf16, WiT, WjT, P, Q);

    // 4) Edge kernel (one 16-edge tile per block)
    edge_kernel<<<n_edges / 16, 256, 0, stream>>>(edge_attr, edge_index, b_ij,
                                                  P, Q, WijT, out, n_edges);
}
